// MultiHeadAttention_21174188769589
// MI455X (gfx1250) — compile-verified, matrix operand order checked by reading, not run on hardware
//
#include <hip/hip_runtime.h>
#include <hip/hip_bf16.h>

// Problem constants (match reference)
#define DM 1024          // d_model
#define BB 4             // batch
#define TT 2048          // seq len
#define NH 16            // heads
#define DK 64            // head dim
#define MTOT (BB*TT)     // 8192 rows

typedef __attribute__((ext_vector_type(16))) _Float16 v16h;
typedef __attribute__((ext_vector_type(8)))  _Float16 v8h;
typedef __attribute__((ext_vector_type(8)))  float    v8f;
typedef __attribute__((ext_vector_type(4)))  float    v4f;

#define AS3 __attribute__((address_space(3)))

static __device__ __forceinline__ v8f wmma32(v16h a, v16h b, v8f c) {
  // D = A(16x32 f16) * B(32x16 f16) + C(16x16 f32)
  return __builtin_amdgcn_wmma_f32_16x16x32_f16(false, a, false, b, (short)0, c,
                                                false, false);
}

// A/B fragment loader, f16 source, row-major, leading dim `ld` (halfs).
// ISA 16-bit 16x32 layout: lane L -> row = L&15, half = L>>4;
// elements 0..7  : k = half*8 + e        (one b128 load)
// elements 8..15 : k = 16 + half*8 + e-8 (one b128 load)
static __device__ __forceinline__ v16h frag16(const _Float16* p0, int ld) {
  const int lane = threadIdx.x & 31;
  const _Float16* p = p0 + (size_t)(lane & 15) * ld + ((lane >> 4) * 8);
  v8h lo = *(const v8h*)p;
  v8h hi = *(const v8h*)(p + 16);
  v16h r;
#pragma unroll
  for (int i = 0; i < 8; ++i) { r[i] = lo[i]; r[i + 8] = hi[i]; }
  return r;
}

// Same fragment, fp32 source with on-the-fly f16 conversion.
static __device__ __forceinline__ v16h frag32(const float* p0, int ld) {
  const int lane = threadIdx.x & 31;
  const float* p = p0 + (size_t)(lane & 15) * ld + ((lane >> 4) * 8);
  v4f x0 = *(const v4f*)(p);
  v4f x1 = *(const v4f*)(p + 4);
  v4f x2 = *(const v4f*)(p + 16);
  v4f x3 = *(const v4f*)(p + 20);
  v16h r;
#pragma unroll
  for (int i = 0; i < 4; ++i) {
    r[i]      = (_Float16)x0[i];
    r[4 + i]  = (_Float16)x1[i];
    r[8 + i]  = (_Float16)x2[i];
    r[12 + i] = (_Float16)x3[i];
  }
  return r;
}

// fp32 -> f16 conversion (weights)
__global__ void cvt_f32_f16(const float* __restrict__ src,
                            _Float16* __restrict__ dst, int n) {
  int i = blockIdx.x * blockDim.x + threadIdx.x;
  int stride = gridDim.x * blockDim.x;
  for (; i < n; i += stride) dst[i] = (_Float16)src[i];
}

// C[M=8192, N=1024] = A[M,1024] @ W^T + bias   (W stored [N,K] row-major, f16)
// Wave computes a 32(m) x 64(n) tile = 2x4 WMMA accumulators.
// All 8 waves of a block share the same 32 A-rows: the 32x32 A k-slice is
// staged into LDS with gfx1250 async copies (ASYNCcnt), double-buffered so the
// DMA for k-step i+1 overlaps the WMMA work of k-step i.
// MODE 0: A fp32, store f16 scaled by 1/8 to Q head-major  [B,H,T,Dk]
// MODE 1: A fp32, store f16 to K head-major                [B,H,T,Dk]
// MODE 2: A fp32, store f16 to V head-major TRANSPOSED     [B,H,Dk,T]
// MODE 3: A f16 (context), store fp32 row-major            [M, DM]
template <int MODE>
__global__ __launch_bounds__(256) void gemm_kernel(const void* __restrict__ Av,
                                                   const _Float16* __restrict__ W,
                                                   const float* __restrict__ bias,
                                                   void* __restrict__ Cv) {
  __shared__ __align__(16) char asmem[2][32 * 32 * 4]; // double buffer, fp32 worst case
  const int tid = threadIdx.x;
  const int wid = blockIdx.x * 8 + (tid >> 5); // 4096 waves
  const int mtile = wid >> 4;   // 0..255 (M/32); uniform within a block
  const int ntile = wid & 15;   // 0..15  (N/64)
  const int m0 = mtile * 32, n0 = ntile * 64;
  const int lane = tid & 31;
  const int hf = lane >> 4, nl = lane & 15;

  const unsigned lds0 = (unsigned)(size_t)(AS3 char*)&asmem[0][0];
  const unsigned lds1 = (unsigned)(size_t)(AS3 char*)&asmem[1][0];

  // one async b128 copy instruction per wave per k-step
  auto issue_copy = [&](unsigned ldsb, int k0) {
    if constexpr (MODE < 3) {
      const float* A = (const float*)Av;
      const int arow = tid >> 3;            // 0..31
      const float* gsrc = A + (size_t)(m0 + arow) * DM + k0 + (tid & 7) * 4;
      const unsigned ldst = ldsb + (unsigned)(arow * 128 + (tid & 7) * 16);
      asm volatile("global_load_async_to_lds_b128 %0, %1, off"
                   :: "v"(ldst), "v"(gsrc) : "memory");
    } else {
      const _Float16* A = (const _Float16*)Av;
      if (tid < 128) {
        const int arow = tid >> 2;          // 0..31
        const _Float16* gsrc = A + (size_t)(m0 + arow) * DM + k0 + (tid & 3) * 8;
        const unsigned ldst = ldsb + (unsigned)(arow * 64 + (tid & 3) * 16);
        asm volatile("global_load_async_to_lds_b128 %0, %1, off"
                     :: "v"(ldst), "v"(gsrc) : "memory");
      }
    }
  };

  v8f acc[2][4] = {};

  issue_copy(lds0, 0); // prologue: stage first A slice

  for (int ki = 0; ki < DM / 32; ++ki) {
    const int k0 = ki * 32;
    const int cur = ki & 1;
    const bool more = (ki + 1) < (DM / 32);
    if (more) issue_copy(cur ? lds0 : lds1, k0 + 32); // overlap next DMA

    // B fragments from global while DMAs are in flight
    v16h b0 = frag16(W + (size_t)(n0 + 0)  * DM + k0, DM);
    v16h b1 = frag16(W + (size_t)(n0 + 16) * DM + k0, DM);
    v16h b2 = frag16(W + (size_t)(n0 + 32) * DM + k0, DM);
    v16h b3 = frag16(W + (size_t)(n0 + 48) * DM + k0, DM);
    if (more) __builtin_prefetch(W + (size_t)n0 * DM + k0 + 32, 0, 1);

    // wait for the PREVIOUS copy only (newest may stay in flight)
    if (more) asm volatile("s_wait_asynccnt 0x1" ::: "memory");
    else      asm volatile("s_wait_asynccnt 0x0" ::: "memory");
    __syncthreads(); // staged A slice visible to whole block

    v16h a0, a1;
    if constexpr (MODE < 3) {
      const float* As = (const float*)&asmem[cur][0];
      a0 = frag32(As, 32);
      a1 = frag32(As + 16 * 32, 32);
    } else {
      const _Float16* Ah = (const _Float16*)&asmem[cur][0];
      a0 = frag16(Ah, 32);
      a1 = frag16(Ah + 16 * 32, 32);
    }
#pragma unroll
    for (int j = 0; j < 4; ++j) {
      const v16h bj = (j == 0) ? b0 : (j == 1) ? b1 : (j == 2) ? b2 : b3;
      acc[0][j] = wmma32(a0, bj, acc[0][j]);
      acc[1][j] = wmma32(a1, bj, acc[1][j]);
    }
    __syncthreads(); // all waves done reading buf[cur] before it is re-targeted
  }

#pragma unroll
  for (int j = 0; j < 4; ++j) {
    const int n = n0 + j * 16 + nl;
    const float bn = bias[n];
#pragma unroll
    for (int i = 0; i < 2; ++i) {
#pragma unroll
      for (int v = 0; v < 8; ++v) {
        const int m = m0 + i * 16 + v + 8 * hf;    // C frag: lane=n, vgpr v -> m
        const float val = acc[i][j][v] + bn;
        if constexpr (MODE == 3) {
          ((float*)Cv)[(size_t)m * DM + n] = val;
        } else {
          const int b = m >> 11, t = m & (TT - 1);
          const int h = n >> 6, d = n & (DK - 1);
          _Float16* C = (_Float16*)Cv;
          if constexpr (MODE == 0)
            C[(((size_t)(b * NH + h)) * TT + t) * DK + d] =
                (_Float16)(val * 0.125f);                // fold 1/sqrt(Dk)
          else if constexpr (MODE == 1)
            C[(((size_t)(b * NH + h)) * TT + t) * DK + d] = (_Float16)val;
          else
            C[(((size_t)(b * NH + h)) * DK + d) * TT + t] = (_Float16)val;
        }
      }
    }
  }
}

// Flash-style causal attention, TRANSPOSED formulation: one wave per
// (b, h, 16-query tile). Computes S^T = K @ Q^T so each lane owns one query
// column; softmax state is per-lane scalar, the P^T fragment for
// O^T = V^T @ P^T is a pure in-lane register repack (no LDS round trip).
__global__ __launch_bounds__(256) void attn_kernel(const _Float16* __restrict__ qh,
                                                   const _Float16* __restrict__ kh,
                                                   const _Float16* __restrict__ vT,
                                                   _Float16* __restrict__ ctx) {
  const int wid = blockIdx.x * 8 + (threadIdx.x >> 5); // 8192 waves
  const int bh = wid >> 7;
  const int q0 = (wid & 127) * 16;
  const int b = bh >> 4, h = bh & 15;
  const int lane = threadIdx.x & 31;
  const int hf = lane >> 4, nl = lane & 15;

  const _Float16* Q = qh + ((size_t)bh * TT + q0) * DK;
  const _Float16* K = kh + (size_t)bh * TT * DK;
  const _Float16* V = vT + (size_t)bh * DK * TT;

  // Q as B-fragments (lane = query column), reused across all key chunks
  const v16h bq0 = frag16(Q, DK);       // d = 0..31
  const v16h bq1 = frag16(Q + 32, DK);  // d = 32..63

  float mrow = -1e30f, lrow = 0.0f;     // per-lane scalars (lane = one q row)
  v8f o[4] = {};                        // O^T: lane = q, elem v -> d = v+8*hf

  auto chunk = [&](int k0, bool masked) {
    // ---- S^T tiles: rows = keys, cols = queries; contraction over Dk=64
    v8f s0 = {}, s1 = {};
    {
      v16h ak;
      ak = frag16(K + (size_t)k0 * DK, DK);             s0 = wmma32(ak, bq0, s0);
      ak = frag16(K + (size_t)k0 * DK + 32, DK);        s0 = wmma32(ak, bq1, s0);
      ak = frag16(K + (size_t)(k0 + 16) * DK, DK);      s1 = wmma32(ak, bq0, s1);
      ak = frag16(K + (size_t)(k0 + 16) * DK + 32, DK); s1 = wmma32(ak, bq1, s1);
    }
    const int q = q0 + nl; // this lane's query index
    if (masked) {
#pragma unroll
      for (int v = 0; v < 8; ++v) {
        if (k0 + v + 8 * hf > q)      s0[v] = -1e30f;  // s0: keys k0 + v+8hf
        if (k0 + 16 + v + 8 * hf > q) s1[v] = -1e30f;  // s1: keys k0+16 + v+8hf
      }
    }
    // ---- per-query max: 16 local values + one half-swap exchange
    float mx = fmaxf(s0[0], s1[0]);
#pragma unroll
    for (int v = 1; v < 8; ++v) mx = fmaxf(mx, fmaxf(s0[v], s1[v]));
    mx = fmaxf(mx, __shfl_xor(mx, 16, 32));
    const float nm = fmaxf(mrow, mx);
    const float sc = __expf(mrow - nm);
    mrow = nm;
    // ---- exponentiate; build P^T B-fragment directly in registers.
    // B-frag lane (q, hf) wants keys {8hf+e} and {16+8hf+e}: exactly s0/s1.
    v16h pfrag;
    float rsl = 0.0f;
#pragma unroll
    for (int v = 0; v < 8; ++v) {
      const float p0 = __expf(s0[v] - nm);
      const float p1 = __expf(s1[v] - nm);
      rsl += p0 + p1;
      pfrag[v]     = (_Float16)p0;
      pfrag[v + 8] = (_Float16)p1;
    }
    lrow = lrow * sc + (rsl + __shfl_xor(rsl, 16, 32));
    // ---- rescale and accumulate O^T += V^T @ P^T (4 d-tiles of 16)
#pragma unroll
    for (int j = 0; j < 4; ++j) {
#pragma unroll
      for (int v = 0; v < 8; ++v) o[j][v] *= sc;
      v16h av = frag16(V + (size_t)(j * 16) * TT + k0, TT);
      o[j] = wmma32(av, pfrag, o[j]);
    }
  };

  int k0 = 0;
  for (; k0 + 32 <= q0; k0 += 32) chunk(k0, false); // fully visible chunks
  for (; k0 < q0 + 16; k0 += 32) chunk(k0, true);   // diagonal (masked) tail

  // ---- normalize; contiguous packed b128 stores into ctx [B,T,DM]
  const float inv = 1.0f / lrow;
  _Float16* crow = ctx + ((size_t)(b * TT + q0 + nl)) * DM + h * DK + 8 * hf;
#pragma unroll
  for (int j = 0; j < 4; ++j) {
    v8h pk;
#pragma unroll
    for (int v = 0; v < 8; ++v) pk[v] = (_Float16)(o[j][v] * inv);
    *(v8h*)(crow + j * 16) = pk; // d = h*64 + j*16 + 8*hf + v, v contiguous
  }
}

extern "C" void kernel_launch(void* const* d_in, const int* in_sizes, int n_in,
                              void* d_out, int out_size, void* d_ws, size_t ws_size,
                              hipStream_t stream) {
  (void)in_sizes; (void)n_in; (void)out_size; (void)ws_size;
  const float* query = (const float*)d_in[0];
  const float* key_i = (const float*)d_in[1];
  const float* value = (const float*)d_in[2];
  // d_in[3] = causal mask, handled analytically
  const float* Wq = (const float*)d_in[4];
  const float* bq = (const float*)d_in[5];
  const float* Wk = (const float*)d_in[6];
  const float* bk = (const float*)d_in[7];
  const float* Wv = (const float*)d_in[8];
  const float* bv = (const float*)d_in[9];
  const float* Wo = (const float*)d_in[10];
  const float* bo = (const float*)d_in[11];

  char* ws = (char*)d_ws;
  size_t off = 0;
  auto alloc = [&](size_t bytes) -> void* {
    void* p = ws + off;
    off += (bytes + 255) & ~(size_t)255;
    return p;
  };
  _Float16* wq16 = (_Float16*)alloc((size_t)DM * DM * 2);
  _Float16* wk16 = (_Float16*)alloc((size_t)DM * DM * 2);
  _Float16* wv16 = (_Float16*)alloc((size_t)DM * DM * 2);
  _Float16* wo16 = (_Float16*)alloc((size_t)DM * DM * 2);
  _Float16* qh   = (_Float16*)alloc((size_t)BB * NH * TT * DK * 2);
  _Float16* kh   = (_Float16*)alloc((size_t)BB * NH * TT * DK * 2);
  _Float16* vT   = (_Float16*)alloc((size_t)BB * NH * DK * TT * 2);
  _Float16* ctx  = (_Float16*)alloc((size_t)MTOT * DM * 2);

  // 1) weights fp32 -> f16
  cvt_f32_f16<<<2048, 256, 0, stream>>>(Wq, wq16, DM * DM);
  cvt_f32_f16<<<2048, 256, 0, stream>>>(Wk, wk16, DM * DM);
  cvt_f32_f16<<<2048, 256, 0, stream>>>(Wv, wv16, DM * DM);
  cvt_f32_f16<<<2048, 256, 0, stream>>>(Wo, wo16, DM * DM);

  // 2) projections (WMMA GEMMs), head-major outputs; Q pre-scaled by 1/8
  gemm_kernel<0><<<512, 256, 0, stream>>>(query, wq16, bq, qh);
  gemm_kernel<1><<<512, 256, 0, stream>>>(key_i, wk16, bk, kh);
  gemm_kernel<2><<<512, 256, 0, stream>>>(value, wv16, bv, vT);

  // 3) causal flash attention (transposed formulation)
  attn_kernel<<<1024, 256, 0, stream>>>(qh, kh, vT, ctx);

  // 4) output projection -> fp32 d_out
  gemm_kernel<3><<<512, 256, 0, stream>>>(ctx, wo16, bo, d_out);
}
